// RWKV_TimeMix_41704132444298
// MI455X (gfx1250) — matrix operand order repaired and checked by reading, not compiled, not run on hardware
//
#include <hip/hip_runtime.h>
#include <hip/hip_bf16.h>

// ---------------------------------------------------------------------------
// RWKV TimeMix forward for MI455X (gfx1250, wave32, WMMA + async LDS staging).
//   T = 767 (padded to 768), C = D = 768.
// Pipeline:
//   1) cvt weights fp32 -> bf16 (single fused launch, blockIdx.y selects)
//   2) mix kernel: xk/xv/xr (bf16, padded row 767 = 0)
//   3) fused k/v/r bf16 WMMA GEMM launch (blockIdx.y selects triple):
//      64x128 block tile staged to LDS via GLOBAL_LOAD_ASYNC_TO_LDS_B128
//      (double buffered, ASYNCcnt), 2x4 waves of 2x2 wmma each
//   4) blocked scan (16 chunks x 48): exp/kv + local scans + carries
//   5) carry-apply + sigmoid + divide -> y (bf16)
//   6) bf16 WMMA GEMM y @ Wo^T -> d_out (fp32, row-guarded to 767)
// ---------------------------------------------------------------------------

typedef __bf16 bf16;
typedef __bf16 bf16x16 __attribute__((ext_vector_type(16)));
typedef __bf16 bf16x8  __attribute__((ext_vector_type(8)));
typedef float  f32x8   __attribute__((ext_vector_type(8)));

#define CDIM   768
#define TLEN   767
#define TPAD   768
#define NKSTEP 24            // 768 / 32 K-steps
#define MBLK   64            // block tile M
#define NBLK   128           // block tile N
#define NBLKS_M (TPAD / MBLK)   // 12
#define NBLKS_N (CDIM / NBLK)   // 6
#define LSTR   40            // padded LDS row stride in bf16 (80 B = 20 banks)
#define ABUF   (MBLK * LSTR)            // 2560 elem
#define BBUF   (NBLK * LSTR)            // 5120 elem
#define BUFSZ  (ABUF + BBUF)            // 7680 elem (15360 B)
#define NCHUNK 16
#define CHUNK  48            // 16 * 48 = 768 >= 767

union FragBF {
    bf16x16 v;
    bf16x8  h[2];
};

static __device__ __forceinline__ f32x8 wmma_bf16(const FragBF& a, const FragBF& b,
                                                  f32x8 c) {
    return __builtin_amdgcn_wmma_f32_16x16x32_bf16(
        /*neg_a=*/false, a.v, /*neg_b=*/false, b.v,
        /*c_mod=*/(short)0, c, /*reuse_a=*/false, /*reuse_b=*/false);
}

// Wave-relative LDS byte offset = low 32 bits of the flat address (ISA 10.2).
static __device__ __forceinline__ unsigned lds_off_u32(const void* p) {
    return (unsigned)(unsigned long long)p;
}

// Issue the async global->LDS staging for one K-step (3 b128 ops / thread):
// A tile 64x32 (1 op) + B tile 128x32 (2 ops). Tracked by ASYNCcnt.
static __device__ __forceinline__ void stage_tiles(const bf16* __restrict__ Ag,
                                                   const bf16* __restrict__ Wg,
                                                   int k0, bf16* sbuf, int tid) {
    int row   = tid >> 2;       // 0..63
    int chunk = tid & 3;        // four 16-B chunks cover 32 bf16 of K
    {
        const bf16* g = Ag + (size_t)row * CDIM + k0 + chunk * 8;
        unsigned l = lds_off_u32(sbuf + row * LSTR + chunk * 8);
        asm volatile("global_load_async_to_lds_b128 %0, %1, off"
                     :: "v"(l), "v"(g) : "memory");
    }
    bf16* sB = sbuf + ABUF;
    {
        const bf16* g0 = Wg + (size_t)row * CDIM + k0 + chunk * 8;
        const bf16* g1 = g0 + (size_t)64 * CDIM;
        unsigned l0 = lds_off_u32(sB + row * LSTR + chunk * 8);
        unsigned l1 = lds_off_u32(sB + (row + 64) * LSTR + chunk * 8);
        asm volatile("global_load_async_to_lds_b128 %0, %1, off"
                     :: "v"(l0), "v"(g0) : "memory");
        asm volatile("global_load_async_to_lds_b128 %0, %1, off"
                     :: "v"(l1), "v"(g1) : "memory");
    }
}

#define WAIT_ASYNC0() asm volatile("s_wait_asynccnt 0x0" ::: "memory")

// Consume one staged K-step from LDS: 8 ds_load_b128 -> 4 wmma.
// Fragment variables are scoped here so their live ranges end before the
// next body's loads (keeps accumulators pinned, avoids acc eviction movs).
static __device__ __forceinline__ void consume_step(const bf16* __restrict__ sbuf,
                                                    int waveM, int waveN,
                                                    int half, int l16,
                                                    f32x8& acc00, f32x8& acc01,
                                                    f32x8& acc10, f32x8& acc11) {
    const bf16* sArow0 = sbuf + (waveM * 32 + l16) * LSTR;
    const bf16* sArow1 = sArow0 + 16 * LSTR;
    const bf16* sBrow0 = sbuf + ABUF + (waveN * 32 + l16) * LSTR;
    const bf16* sBrow1 = sBrow0 + 16 * LSTR;

    FragBF a0, a1, b0, b1;
    // A 16x32 layout: lane(half,m) holds K = {8*half+0..7, 16+8*half+0..7}
    a0.h[0] = *(const bf16x8*)(sArow0 + half * 8);
    a0.h[1] = *(const bf16x8*)(sArow0 + 16 + half * 8);
    a1.h[0] = *(const bf16x8*)(sArow1 + half * 8);
    a1.h[1] = *(const bf16x8*)(sArow1 + 16 + half * 8);
    // B 32x16 layout: lane(half,n) holds K = {8*half+0..7, 16+8*half+0..7} of column n
    b0.h[0] = *(const bf16x8*)(sBrow0 + half * 8);
    b0.h[1] = *(const bf16x8*)(sBrow0 + 16 + half * 8);
    b1.h[0] = *(const bf16x8*)(sBrow1 + half * 8);
    b1.h[1] = *(const bf16x8*)(sBrow1 + 16 + half * 8);

    acc00 = wmma_bf16(a0, b0, acc00);
    acc01 = wmma_bf16(a0, b1, acc01);
    acc10 = wmma_bf16(a1, b0, acc10);
    acc11 = wmma_bf16(a1, b1, acc11);
}

// ---------------------------------------------------------------------------
// Block GEMM body: D[m,n] = sum_k A[m,k] * W[n,k]  (A, W row-major bf16).
// One 256-thread block computes a 64x128 tile; waves in a 2x4 grid each own
// a 32x32 (2x2 wmma) sub-tile. A/B staged to double-buffered LDS with async
// DMA overlapping the ds_load + wmma consumption of the previous stage.
// Manual two-body unroll: buffer bases are compile-time offsets.
// ---------------------------------------------------------------------------
template <bool GUARD>
static __device__ __forceinline__ void gemm_lds_body(const bf16* __restrict__ A,
                                                     const bf16* __restrict__ W,
                                                     float* __restrict__ D,
                                                     int mValid) {
    __shared__ bf16 smem[2 * BUFSZ];

    int tid  = threadIdx.x;
    int wave = tid >> 5;
    int lane = tid & 31;
    int mb = blockIdx.x / NBLKS_N;
    int nb = blockIdx.x - mb * NBLKS_N;

    const bf16* Ag = A + (size_t)mb * MBLK * CDIM;
    const bf16* Wg = W + (size_t)nb * NBLK * CDIM;

    int waveM = wave >> 2;       // 0..1
    int waveN = wave & 3;        // 0..3
    int half  = lane >> 4;       // 0 or 1
    int l16   = lane & 15;

    f32x8 acc00 = {}, acc01 = {}, acc10 = {}, acc11 = {};

    stage_tiles(Ag, Wg, 0, smem, tid);
#pragma unroll 1
    for (int i = 0; i < NKSTEP; i += 2) {
        // ---- body 0: consume buf0, prefetch (i+1) into buf1 ----
        WAIT_ASYNC0();           // own stage-i async ops complete
        __syncthreads();         // all waves staged; prev consume finished
        if (i + 1 < NKSTEP)
            stage_tiles(Ag, Wg, (i + 1) * 32, smem + BUFSZ, tid);
        consume_step(smem, waveM, waveN, half, l16, acc00, acc01, acc10, acc11);

        // ---- body 1: consume buf1, prefetch (i+2) into buf0 ----
        WAIT_ASYNC0();
        __syncthreads();
        if (i + 2 < NKSTEP)
            stage_tiles(Ag, Wg, (i + 2) * 32, smem, tid);
        consume_step(smem + BUFSZ, waveM, waveN, half, l16,
                     acc00, acc01, acc10, acc11);
    }

    // D layout: element r -> row (8*half + r) of the 16x16 tile, col l16
    int n0 = nb * NBLK + waveN * 32 + l16;
    int n1 = n0 + 16;
    int m0 = mb * MBLK + waveM * 32 + half * 8;
    int m1 = m0 + 16;
#pragma unroll
    for (int r = 0; r < 8; ++r) {
        int ma = m0 + r;
        int mb_ = m1 + r;
        if (!GUARD || ma < mValid) {
            D[(size_t)ma * CDIM + n0] = acc00[r];
            D[(size_t)ma * CDIM + n1] = acc01[r];
        }
        if (!GUARD || mb_ < mValid) {
            D[(size_t)mb_ * CDIM + n0] = acc10[r];
            D[(size_t)mb_ * CDIM + n1] = acc11[r];
        }
    }
}

// Fused k/v/r GEMM launch: blockIdx.y in {0,1,2} selects the triple.
__global__ void __launch_bounds__(256)
__attribute__((amdgpu_waves_per_eu(2)))
gemm_kvr_kernel(const bf16* __restrict__ xk, const bf16* __restrict__ wk,
                float* __restrict__ dk,
                const bf16* __restrict__ xv, const bf16* __restrict__ wv,
                float* __restrict__ dv,
                const bf16* __restrict__ xr, const bf16* __restrict__ wr,
                float* __restrict__ dr) {
    const bf16* A;
    const bf16* W;
    float* D;
    if (blockIdx.y == 0)      { A = xk; W = wk; D = dk; }
    else if (blockIdx.y == 1) { A = xv; W = wv; D = dv; }
    else                      { A = xr; W = wr; D = dr; }
    gemm_lds_body<false>(A, W, D, TPAD);
}

// Final output GEMM: row-guarded to the 767 real rows of d_out.
__global__ void __launch_bounds__(256)
__attribute__((amdgpu_waves_per_eu(2)))
gemm_out_kernel(const bf16* __restrict__ A, const bf16* __restrict__ W,
                float* __restrict__ D, int mValid) {
    gemm_lds_body<true>(A, W, D, mValid);
}

// ---------------------------------------------------------------------------
// fp32 -> bf16 conversion of all 4 weight matrices in one launch
// ---------------------------------------------------------------------------
__global__ void cvt4_bf16_kernel(const float* __restrict__ s0, bf16* __restrict__ d0,
                                 const float* __restrict__ s1, bf16* __restrict__ d1,
                                 const float* __restrict__ s2, bf16* __restrict__ d2,
                                 const float* __restrict__ s3, bf16* __restrict__ d3,
                                 int n) {
    int i = blockIdx.x * blockDim.x + threadIdx.x;
    if (i >= n) return;
    const float* s;
    bf16* d;
    switch (blockIdx.y) {
        case 0:  s = s0; d = d0; break;
        case 1:  s = s1; d = d1; break;
        case 2:  s = s2; d = d2; break;
        default: s = s3; d = d3; break;
    }
    d[i] = (bf16)s[i];
}

// ---------------------------------------------------------------------------
// time-shift + mix -> bf16 activations (padded to 768 rows, row 767 = 0)
// ---------------------------------------------------------------------------
__global__ void mix_kernel(const float* __restrict__ x,
                           const float* __restrict__ mk,
                           const float* __restrict__ mv,
                           const float* __restrict__ mr,
                           bf16* __restrict__ xk,
                           bf16* __restrict__ xv,
                           bf16* __restrict__ xr) {
    int i = blockIdx.x * blockDim.x + threadIdx.x;   // 0 .. 768*768-1
    if (i >= TPAD * CDIM) return;
    int t = i / CDIM;
    int c = i - t * CDIM;
    float vk = 0.f, vv = 0.f, vr = 0.f;
    if (t < TLEN) {
        float xc = x[i];
        float xp = (t == 0) ? 0.f : x[i - CDIM];
        float a;
        a = mk[c]; vk = xc * a + xp * (1.f - a);
        a = mv[c]; vv = xc * a + xp * (1.f - a);
        a = mr[c]; vr = xc * a + xp * (1.f - a);
    }
    xk[i] = (bf16)vk;
    xv[i] = (bf16)vv;
    xr[i] = (bf16)vr;
}

// ---------------------------------------------------------------------------
// Scan pass 1: per (chunk, channel): exponentiate k, form kv, local scan
// with zero carry-in. kexp overwrites klin, kv overwrites vlin (in place).
// Stores A_local (value BEFORE absorbing element t) and chunk totals S.
// ---------------------------------------------------------------------------
__global__ void scan_pass1(float* __restrict__ klin,      // in: k_lin, out: exp(k)
                           float* __restrict__ vlin,      // in: v_lin, out: kv
                           const float* __restrict__ time_decay,
                           float* __restrict__ aloc_kv,
                           float* __restrict__ aloc_k,
                           float* __restrict__ Skv,
                           float* __restrict__ Sk) {
    int i = blockIdx.x * blockDim.x + threadIdx.x;  // 16*768 threads
    if (i >= NCHUNK * CDIM) return;
    int d     = i % CDIM;
    int chunk = i / CDIM;
    int t0 = chunk * CHUNK;
    int L  = (TLEN - t0 < CHUNK) ? (TLEN - t0) : CHUNK;

    float dec  = __expf(time_decay[d]);
    float wdec = __expf(-dec);

    float Akv = 0.f, Ak = 0.f;
    for (int s = 0; s < L; ++s) {
        size_t idx = (size_t)(t0 + s) * CDIM + d;
        float kl = klin[idx];
        float vl = vlin[idx];
        float kk = __expf(fminf(kl, 60.f));
        float kv = kk * vl;
        klin[idx] = kk;
        vlin[idx] = kv;
        aloc_kv[idx] = Akv;
        aloc_k[idx]  = Ak;
        Akv = wdec * Akv + kv;
        Ak  = wdec * Ak + kk;
    }
    Skv[chunk * CDIM + d] = Akv;
    Sk[chunk * CDIM + d]  = Ak;
}

// ---------------------------------------------------------------------------
// Scan pass 2: per channel, serial exclusive scan over 16 chunk totals.
// carry[c+1] = w^{L_c} * carry[c] + S[c]
// ---------------------------------------------------------------------------
__global__ void scan_pass2(const float* __restrict__ time_decay,
                           const float* __restrict__ Skv,
                           const float* __restrict__ Sk,
                           float* __restrict__ Ckv,
                           float* __restrict__ Ck) {
    int d = blockIdx.x * blockDim.x + threadIdx.x;
    if (d >= CDIM) return;
    float dec = __expf(time_decay[d]);
    float ckv = 0.f, ck = 0.f;
#pragma unroll
    for (int c = 0; c < NCHUNK; ++c) {
        Ckv[c * CDIM + d] = ckv;
        Ck[c * CDIM + d]  = ck;
        int t0 = c * CHUNK;
        int L  = (TLEN - t0 < CHUNK) ? (TLEN - t0) : CHUNK;
        float wL = __expf(-dec * (float)L);
        ckv = wL * ckv + Skv[c * CDIM + d];
        ck  = wL * ck  + Sk[c * CDIM + d];
    }
}

// ---------------------------------------------------------------------------
// Scan pass 3: apply carries, finish wkv/wk, sigmoid(r) * wkv/wk -> y (bf16)
// ---------------------------------------------------------------------------
__global__ void scan_pass3(const float* __restrict__ kexp,
                           const float* __restrict__ kv,
                           const float* __restrict__ rlin,
                           const float* __restrict__ aloc_kv,
                           const float* __restrict__ aloc_k,
                           const float* __restrict__ Ckv,
                           const float* __restrict__ Ck,
                           const float* __restrict__ time_decay,
                           const float* __restrict__ time_first,
                           bf16* __restrict__ y) {
    int i = blockIdx.x * blockDim.x + threadIdx.x;   // 768*768
    if (i >= TPAD * CDIM) return;
    int t = i / CDIM;
    int d = i - t * CDIM;
    if (t >= TLEN) { y[i] = (bf16)0.f; return; }

    int chunk = t / CHUNK;
    int s     = t - chunk * CHUNK;

    float dec  = __expf(time_decay[d]);
    float fade = __expf(-dec * (float)s);            // w^s
    float tf   = __expf(time_first[d]);

    float wkv = tf * kv[i]   + aloc_kv[i] + Ckv[chunk * CDIM + d] * fade;
    float wk  = tf * kexp[i] + aloc_k[i]  + Ck[chunk * CDIM + d]  * fade + 1e-8f;
    float sig = 1.f / (1.f + __expf(-rlin[i]));
    y[i] = (bf16)(sig * wkv / wk);
}

// ---------------------------------------------------------------------------
extern "C" void kernel_launch(void* const* d_in, const int* in_sizes, int n_in,
                              void* d_out, int out_size, void* d_ws, size_t ws_size,
                              hipStream_t stream) {
    const float* x    = (const float*)d_in[0];
    const float* tdec = (const float*)d_in[1];
    const float* tfst = (const float*)d_in[2];
    const float* mk   = (const float*)d_in[3];
    const float* mv   = (const float*)d_in[4];
    const float* mr   = (const float*)d_in[5];
    const float* Wk   = (const float*)d_in[6];
    const float* Wv   = (const float*)d_in[7];
    const float* Wr   = (const float*)d_in[8];
    const float* Wo   = (const float*)d_in[9];
    float* out = (float*)d_out;

    const size_t S = (size_t)TPAD * CDIM;   // 589824
    char* p = (char*)d_ws;
    auto carve = [&](size_t bytes) {
        void* r = (void*)p;
        p += (bytes + 255) & ~(size_t)255;
        return r;
    };

    bf16* wkb = (bf16*)carve(S * sizeof(bf16));
    bf16* wvb = (bf16*)carve(S * sizeof(bf16));
    bf16* wrb = (bf16*)carve(S * sizeof(bf16));
    bf16* wob = (bf16*)carve(S * sizeof(bf16));
    bf16* xkb = (bf16*)carve(S * sizeof(bf16));
    bf16* xvb = (bf16*)carve(S * sizeof(bf16));
    bf16* xrb = (bf16*)carve(S * sizeof(bf16));
    float* klin = (float*)carve(S * sizeof(float));   // becomes exp(k)
    float* vlin = (float*)carve(S * sizeof(float));   // becomes kv
    float* rlin = (float*)carve(S * sizeof(float));
    float* aloc_kv = (float*)carve(S * sizeof(float));
    float* aloc_k  = (float*)carve(S * sizeof(float));
    float* Skv = (float*)carve((size_t)NCHUNK * CDIM * sizeof(float));
    float* Sk  = (float*)carve((size_t)NCHUNK * CDIM * sizeof(float));
    float* Ckv = (float*)carve((size_t)NCHUNK * CDIM * sizeof(float));
    float* Ck  = (float*)carve((size_t)NCHUNK * CDIM * sizeof(float));
    bf16* ybf = (bf16*)carve(S * sizeof(bf16));

    const int BT = 256;
    const int nElem = (int)S;                 // 589824
    const int gElem = nElem / BT;             // 2304

    // 1) weights -> bf16 (one launch, 4 weight matrices in y)
    cvt4_bf16_kernel<<<dim3(gElem, 4), BT, 0, stream>>>(
        Wk, wkb, Wv, wvb, Wr, wrb, Wo, wob, nElem);

    // 2) mixes
    mix_kernel<<<gElem, BT, 0, stream>>>(x, mk, mv, mr, xkb, xvb, xrb);

    // 3) fused k/v/r WMMA GEMMs: 12x6 block tiles of 64x128, x3 GEMMs in y
    const int gGemm = NBLKS_M * NBLKS_N;      // 72
    gemm_kvr_kernel<<<dim3(gGemm, 3), BT, 0, stream>>>(
        xkb, wkb, klin, xvb, wvb, vlin, xrb, wrb, rlin);

    // 4) blocked scan
    scan_pass1<<<(NCHUNK * CDIM) / BT, BT, 0, stream>>>(
        klin, vlin, tdec, aloc_kv, aloc_k, Skv, Sk);
    scan_pass2<<<CDIM / BT, BT, 0, stream>>>(tdec, Skv, Sk, Ckv, Ck);
    scan_pass3<<<gElem, BT, 0, stream>>>(
        klin, vlin, rlin, aloc_kv, aloc_k, Ckv, Ck, tdec, tfst, ybf);

    // 5) output GEMM, guarded to the 767 real rows of d_out
    gemm_out_kernel<<<gGemm, BT, 0, stream>>>(ybf, wob, out, TLEN);
}
